// SemanticGuidedOT_89008902242758
// MI455X (gfx1250) — matrix operand order repaired and checked by reading, not compiled, not run on hardware
//
#include <hip/hip_runtime.h>

#define BB 8
#define CC 256
#define HW 48
#define NPIX 2304
#define TDIM 512
#define KHEAD 2304
#define OCH 128
#define EPS_OT 0.05f
#define INV_EPS 20.0f

typedef __attribute__((ext_vector_type(16))) _Float16 v16h;
typedef __attribute__((ext_vector_type(16))) __bf16  v16bf;
typedef __attribute__((ext_vector_type(8)))  float   v8f;

union U16F { uint4 u[2]; v16h h; v16bf b; };

__device__ __forceinline__ unsigned short f2bf(float x){
  unsigned int u = __float_as_uint(x);
  u = (u + 0x7FFFu + ((u >> 16) & 1u)) >> 16;
  return (unsigned short)u;
}

// A fragment (16x32, 16-bit): lane L%16 = row M, hi half selects K {0..7,16..23} vs {8..15,24..31}
__device__ __forceinline__ void ldA(const unsigned short* base, int ld, int r0, int k0, int lane, U16F& f){
  int hl = lane & 15, hb = lane >> 4;
  const unsigned short* p = base + (size_t)(r0 + hl) * ld + k0 + hb * 8;
  f.u[0] = *(const uint4*)(p);
  f.u[1] = *(const uint4*)(p + 16);
}
// B fragment (32x16, 16-bit): lane L%16 = output column (= row of row-major B^T source), hi half selects K {0..7,16..23} vs {8..15,24..31}
__device__ __forceinline__ void ldB(const unsigned short* base, int ld, int r0, int k0, int lane, U16F& f){
  int hl = lane & 15, hb = lane >> 4;
  const unsigned short* p = base + (size_t)(r0 + hl) * ld + k0 + hb * 8;
  f.u[0] = *(const uint4*)(p);
  f.u[1] = *(const uint4*)(p + 16);
}

// ---------------- tiny text pipeline: emb, avec = wo(wv emb + bv)+bo, txt_n ----------------
__global__ void k_text(const float* __restrict__ txt, const float* __restrict__ tpw,
                       const float* __restrict__ tpb, const float* __restrict__ wv,
                       const float* __restrict__ bv, const float* __restrict__ wo,
                       const float* __restrict__ bo, float* avec, float* txtn){
  __shared__ float emb[BB*CC];
  __shared__ float vt[BB*CC];
  __shared__ float nr[BB];
  int t = threadIdx.x;
  for (int idx = t; idx < BB*CC; idx += 256){
    int b = idx >> 8, c = idx & 255;
    float s = 0.f; const float* x = txt + b*TDIM; const float* w = tpw + (size_t)c*TDIM;
    for (int j = 0; j < TDIM; ++j) s += x[j]*w[j];
    emb[idx] = s + tpb[c];
  }
  __syncthreads();
  for (int idx = t; idx < BB*CC; idx += 256){
    int b = idx >> 8, c = idx & 255;
    float s = 0.f; const float* e = emb + b*CC; const float* w = wv + (size_t)c*CC;
    for (int j = 0; j < CC; ++j) s += e[j]*w[j];
    vt[idx] = s + bv[c];
  }
  __syncthreads();
  for (int idx = t; idx < BB*CC; idx += 256){
    int b = idx >> 8, c = idx & 255;
    float s = 0.f; const float* e = vt + b*CC; const float* w = wo + (size_t)c*CC;
    for (int j = 0; j < CC; ++j) s += e[j]*w[j];
    avec[idx] = s + bo[c];
  }
  if (t < BB){ float s = 0.f; for (int j = 0; j < CC; ++j){ float v = emb[t*CC+j]; s += v*v; }
               nr[t] = fmaxf(sqrtf(s), 1e-12f); }
  __syncthreads();
  for (int idx = t; idx < BB*CC; idx += 256) txtn[idx] = emb[idx] / nr[idx >> 8];
}

// ---------------- weight conversions to bf16 ----------------
__global__ void k_convw(const float* __restrict__ opw, const float* __restrict__ w1,
                        const float* __restrict__ v1, unsigned short* opwh,
                        unsigned short* w1h, unsigned short* v1h){
  int idx = blockIdx.x * 256 + threadIdx.x;
  if (idx < CC*CC) opwh[idx] = f2bf(opw[idx]);
  if (idx < OCH*KHEAD){ w1h[idx] = f2bf(w1[idx]); v1h[idx] = f2bf(v1[idx]); }
}

// ---------------- source_feat = LN(img + avec); emit sfT(bf16), src_n(f16), cos ----------------
__global__ void k_src(const float* __restrict__ img, const float* __restrict__ avec,
                      const float* __restrict__ lng, const float* __restrict__ lnb,
                      const float* __restrict__ txtn, unsigned short* sft,
                      unsigned short* srch, float* cosb){
  int t = threadIdx.x;
  int b = blockIdx.x / 9;
  int n = (blockIdx.x % 9) * 256 + t;
  const float* ib = img + (size_t)b*CC*NPIX + n;
  const float* av = avec + b*CC;
  float sum = 0.f, sq = 0.f;
  for (int c = 0; c < CC; ++c){ float v = ib[(size_t)c*NPIX] + av[c]; sum += v; sq += v*v; }
  float mean = sum * (1.f/CC);
  float var  = sq  * (1.f/CC) - mean*mean;
  float rstd = rsqrtf(var + 1e-5f);
  float n2 = 0.f;
  for (int c = 0; c < CC; ++c){
    float v = ib[(size_t)c*NPIX] + av[c];
    float y = (v - mean) * rstd * lng[c] + lnb[c];
    n2 += y*y;
    sft[((size_t)b*CC + c)*NPIX + n] = f2bf(y);
  }
  float rn = 1.f / fmaxf(sqrtf(n2), 1e-12f);
  const float* tn = txtn + b*CC;
  float cs = 0.f;
  unsigned short* sh = srch + ((size_t)b*NPIX + n)*CC;
  for (int c = 0; c < CC; ++c){
    float v = ib[(size_t)c*NPIX] + av[c];
    float y = (v - mean) * rstd * lng[c] + lnb[c];
    float z = y * rn;
    union { _Float16 f; unsigned short s; } cv; cv.f = (_Float16)z;
    sh[c] = cv.s;
    cs += z * tn[c];
  }
  cosb[b*NPIX + n] = cs;
}

// ---------------- p / q distributions (log domain), zero Sinkhorn potentials ----------------
__global__ void k_pq(const float* __restrict__ cosb, const float* __restrict__ dens,
                     const float* __restrict__ temp, float* logp, float* logq,
                     float* fb, float* gb){
  __shared__ float red[256];
  int b = blockIdx.x, t = threadIdx.x;
  const float* cb = cosb + b*NPIX;
  float mx = -1e30f;
  for (int n = t; n < NPIX; n += 256) mx = fmaxf(mx, cb[n]);
  red[t] = mx; __syncthreads();
  for (int o = 128; o > 0; o >>= 1){ if (t < o) red[t] = fmaxf(red[t], red[t+o]); __syncthreads(); }
  float maxc = red[0]; __syncthreads();
  float tp = fmaxf(temp[0], 0.01f);
  float s = 0.f;
  for (int n = t; n < NPIX; n += 256){
    float c0 = cb[n];
    s += expf((c0 - maxc) / tp) * (c0 > 0.f ? 1.f : 0.f) + 1e-8f;
  }
  red[t] = s; __syncthreads();
  for (int o = 128; o > 0; o >>= 1){ if (t < o) red[t] += red[t+o]; __syncthreads(); }
  float S = red[0]; __syncthreads();
  for (int n = t; n < NPIX; n += 256){
    float c0 = cb[n];
    float pu = expf((c0 - maxc) / tp) * (c0 > 0.f ? 1.f : 0.f) + 1e-8f;
    logp[b*NPIX + n] = logf(pu / S + 1e-12f);
    fb[b*NPIX + n] = 0.f; gb[b*NPIX + n] = 0.f;
  }
  const float* db = dens + b*NPIX;
  float sq = 0.f;
  for (int n = t; n < NPIX; n += 256) sq += fmaxf(db[n], 0.f) + 1e-6f;
  red[t] = sq; __syncthreads();
  for (int o = 128; o > 0; o >>= 1){ if (t < o) red[t] += red[t+o]; __syncthreads(); }
  float Sq = red[0]; __syncthreads();
  for (int n = t; n < NPIX; n += 256){
    float tq = fmaxf(db[n], 0.f) + 1e-6f;
    logq[b*NPIX + n] = logf(tq / Sq + 1e-12f);
  }
}

// ---------------- cost = 1 - S S^T + sw * spatial   (f16 WMMA) ----------------
__global__ void k_cost(const unsigned short* __restrict__ srch, const float* __restrict__ sw,
                       float* cost){
  int t = threadIdx.x, lane = t & 31, wv = t >> 5;
  int tile = blockIdx.x * 8 + wv;
  int b = tile / 20736; int r = tile % 20736;
  int i0 = (r / 144) * 16, j0 = (r % 144) * 16;
  const unsigned short* S = srch + (size_t)b*NPIX*CC;
  v8f acc = {};
  for (int k0 = 0; k0 < CC; k0 += 32){
    U16F a, bb;
    ldA(S, CC, i0, k0, lane, a);
    ldB(S, CC, j0, k0, lane, bb);
    acc = __builtin_amdgcn_wmma_f32_16x16x32_f16(false, a.h, false, bb.h, (short)0, acc, false, false);
  }
  float w = sw[0];
  int hl = lane & 15, hb = lane >> 4;
  int col = j0 + hl;
  float ym = (float)(col / HW) * (1.f/47.f), xm = (float)(col % HW) * (1.f/47.f);
  float* cr = cost + (size_t)b*NPIX*NPIX;
  #pragma unroll
  for (int rr = 0; rr < 8; ++rr){
    int row = i0 + rr + hb*8;
    float yn = (float)(row / HW) * (1.f/47.f), xn = (float)(row % HW) * (1.f/47.f);
    float dy = yn - ym, dx = xn - xm;
    float sp = sqrtf(dy*dy + dx*dx) * (1.f/1.41421356f);
    cr[(size_t)row*NPIX + col] = 1.f - acc[rr] + w * sp;
  }
}

// ---------------- Sinkhorn row update: f ----------------
__global__ void k_row(const float* __restrict__ cost, const float* __restrict__ gb,
                      const float* __restrict__ logp, float* fb){
  __shared__ float sm[256], ss[256];
  int t = threadIdx.x;
  int b = blockIdx.x / NPIX, n = blockIdx.x % NPIX;
  const float* cr = cost + ((size_t)b*NPIX + n)*NPIX;
  const float* g = gb + b*NPIX;
  float m = -1e30f, s = 0.f;
  for (int mm = t; mm < NPIX; mm += 256){
    float v = (g[mm] - cr[mm]) * INV_EPS;
    if (v > m){ s = s*expf(m - v) + 1.f; m = v; } else s += expf(v - m);
  }
  sm[t] = m; ss[t] = s; __syncthreads();
  for (int o = 128; o > 0; o >>= 1){
    if (t < o){
      float m1 = sm[t], m2 = sm[t+o], s1 = ss[t], s2 = ss[t+o];
      float M = fmaxf(m1, m2);
      sm[t] = M; ss[t] = s1*expf(m1 - M) + s2*expf(m2 - M);
    }
    __syncthreads();
  }
  if (t == 0){
    float lse = sm[0] + logf(ss[0]);
    fb[b*NPIX + n] = EPS_OT * (logp[b*NPIX + n] - lse);
  }
}

// ---------------- Sinkhorn column update: g ----------------
__global__ void k_col(const float* __restrict__ cost, const float* __restrict__ fbuf,
                      const float* __restrict__ logq, float* gb){
  __shared__ float sm[256], ss[256];
  int t = threadIdx.x, cm = t & 63, rg = t >> 6;
  int b = blockIdx.x / 36, m0 = (blockIdx.x % 36) * 64;
  const float* cb = cost + (size_t)b*NPIX*NPIX;
  const float* f = fbuf + b*NPIX;
  float m = -1e30f, s = 0.f;
  for (int n = rg; n < NPIX; n += 4){
    float v = (f[n] - cb[(size_t)n*NPIX + m0 + cm]) * INV_EPS;
    if (v > m){ s = s*expf(m - v) + 1.f; m = v; } else s += expf(v - m);
  }
  sm[t] = m; ss[t] = s; __syncthreads();
  if (rg == 0){
    float M = sm[cm], S = ss[cm];
    for (int r = 1; r < 4; ++r){
      float m2 = sm[r*64 + cm], s2 = ss[r*64 + cm];
      float Mn = fmaxf(M, m2);
      S = S*expf(M - Mn) + s2*expf(m2 - Mn); M = Mn;
    }
    gb[b*NPIX + m0 + cm] = EPS_OT * (logq[b*NPIX + m0 + cm] - (M + logf(S)));
  }
}

// ---------------- T^T (bf16) via LDS 32x32 transpose ----------------
__global__ void k_T(const float* __restrict__ cost, const float* __restrict__ fbuf,
                    const float* __restrict__ gbuf, unsigned short* Tt){
  __shared__ float tile[32][33];
  int t = threadIdx.x;
  int blk = blockIdx.x;
  int b = blk / 5184; int r = blk % 5184;
  int n0 = (r / 72) * 32, m0 = (r % 72) * 32;
  const float* cb = cost + (size_t)b*NPIX*NPIX;
  int cl = t & 31, rw = t >> 5;
  for (int it = 0; it < 4; ++it){
    int row = rw + it*8;
    tile[row][cl] = cb[(size_t)(n0 + row)*NPIX + m0 + cl];
  }
  __syncthreads();
  const float* f = fbuf + b*NPIX; const float* g = gbuf + b*NPIX;
  unsigned short* To = Tt + (size_t)b*NPIX*NPIX;
  for (int it = 0; it < 4; ++it){
    int rm = rw + it*8;
    float v = expf((f[n0 + cl] + g[m0 + rm] - tile[cl][rm]) * INV_EPS);
    To[(size_t)(m0 + rm)*NPIX + n0 + cl] = f2bf(v);
  }
}

// ---------------- fused = T^T @ source_feat   (bf16 WMMA, K = 2304) ----------------
__global__ void k_fused(const unsigned short* __restrict__ Tt, const unsigned short* __restrict__ sft,
                        unsigned short* fused){
  int t = threadIdx.x, lane = t & 31, wv = t >> 5;
  int tile = blockIdx.x * 8 + wv;
  int b = tile / 2304; int r = tile % 2304;
  int m0 = (r / 16) * 16, c0 = (r % 16) * 16;
  const unsigned short* A  = Tt  + (size_t)b*NPIX*NPIX;
  const unsigned short* Bm = sft + (size_t)b*CC*NPIX;
  v8f acc = {};
  for (int k0 = 0; k0 < NPIX; k0 += 32){
    U16F a, bb;
    ldA(A, NPIX, m0, k0, lane, a);
    ldB(Bm, NPIX, c0, k0, lane, bb);
    acc = __builtin_amdgcn_wmma_f32_16x16x32_bf16(false, a.b, false, bb.b, (short)0, acc, false, false);
  }
  int hl = lane & 15, hb = lane >> 4;
  #pragma unroll
  for (int rr = 0; rr < 8; ++rr){
    int m = m0 + rr + hb*8, c = c0 + hl;
    fused[((size_t)b*NPIX + m)*CC + c] = f2bf(acc[rr]);
  }
}

// ---------------- out_feat = img + op_w @ fused + op_b   (bf16 WMMA) ----------------
__global__ void k_out(const unsigned short* __restrict__ fused, const unsigned short* __restrict__ opwh,
                      const float* __restrict__ img, const float* __restrict__ opb,
                      float* outf, unsigned short* outh){
  int t = threadIdx.x, lane = t & 31, wv = t >> 5;
  int tile = blockIdx.x * 8 + wv;
  int b = tile / 2304; int r = tile % 2304;
  int n0 = (r / 16) * 16, co0 = (r % 16) * 16;
  const unsigned short* A = fused + (size_t)b*NPIX*CC;
  v8f acc = {};
  for (int k0 = 0; k0 < CC; k0 += 32){
    U16F a, bb;
    ldA(A, CC, n0, k0, lane, a);
    ldB(opwh, CC, co0, k0, lane, bb);
    acc = __builtin_amdgcn_wmma_f32_16x16x32_bf16(false, a.b, false, bb.b, (short)0, acc, false, false);
  }
  int hl = lane & 15, hb = lane >> 4;
  int co = co0 + hl;
  float bias = opb[co];
  #pragma unroll
  for (int rr = 0; rr < 8; ++rr){
    int n = n0 + rr + hb*8;
    size_t idx = ((size_t)b*CC + co)*NPIX + n;
    float v = acc[rr] + img[idx] + bias;
    outf[idx] = v;
    outh[idx] = f2bf(v);
  }
}

// ---------------- im2col of out_feat (bf16), k = ic*9 + ky*3 + kx ----------------
__global__ void k_im2col(const unsigned short* __restrict__ outh, unsigned short* A2){
  for (int i = 0; i < 8; ++i){
    size_t idx = (size_t)blockIdx.x * 2048 + threadIdx.x + i*256;
    int k = (int)(idx % KHEAD);
    size_t nb = idx / KHEAD;
    int n = (int)(nb % NPIX);
    int b = (int)(nb / NPIX);
    int kx = k % 3, ky = (k / 3) % 3, ic = k / 9;
    int hy = n / HW + ky - 1, wx = n % HW + kx - 1;
    unsigned short v = 0;
    if (hy >= 0 && hy < HW && wx >= 0 && wx < HW)
      v = outh[((size_t)b*CC + ic)*NPIX + hy*HW + wx];
    A2[idx] = v;
  }
}

// ---------------- head conv3x3 as GEMM (bf16 WMMA, K = 2304) ----------------
__global__ void k_headgemm(const unsigned short* __restrict__ A2, const unsigned short* __restrict__ Wh,
                           const float* __restrict__ bias, float* hbuf){
  int t = threadIdx.x, lane = t & 31, wv = t >> 5;
  int tile = blockIdx.x * 8 + wv;
  int b = tile / 1152; int r = tile % 1152;
  int n0 = (r / 8) * 16, o0 = (r % 8) * 16;
  const unsigned short* A = A2 + (size_t)b*NPIX*KHEAD;
  v8f acc = {};
  for (int k0 = 0; k0 < KHEAD; k0 += 32){
    U16F a, bb;
    ldA(A, KHEAD, n0, k0, lane, a);
    ldB(Wh, KHEAD, o0, k0, lane, bb);
    acc = __builtin_amdgcn_wmma_f32_16x16x32_bf16(false, a.b, false, bb.b, (short)0, acc, false, false);
  }
  int hl = lane & 15, hb = lane >> 4;
  int oc = o0 + hl;
  float bi = bias[oc];
  #pragma unroll
  for (int rr = 0; rr < 8; ++rr){
    int n = n0 + rr + hb*8;
    hbuf[((size_t)b*NPIX + n)*OCH + oc] = acc[rr] + bi;
  }
}

// ---------------- BN batch statistics per channel ----------------
__global__ void k_bn(const float* __restrict__ h0, const float* __restrict__ h1, float* stats){
  __shared__ float s1[256], s2[256];
  int t = threadIdx.x;
  int head = blockIdx.x >> 7, oc = blockIdx.x & 127;
  const float* h = head ? h1 : h0;
  float a = 0.f, b2 = 0.f;
  for (int i = t; i < BB*NPIX; i += 256){ float v = h[(size_t)i*OCH + oc]; a += v; b2 += v*v; }
  s1[t] = a; s2[t] = b2; __syncthreads();
  for (int o = 128; o > 0; o >>= 1){ if (t < o){ s1[t] += s1[t+o]; s2[t] += s2[t+o]; } __syncthreads(); }
  if (t == 0){
    float mean = s1[0] / (float)(BB*NPIX);
    float var  = s2[0] / (float)(BB*NPIX) - mean*mean;
    stats[head*256 + oc] = mean;
    stats[head*256 + 128 + oc] = var;
  }
}

// ---------------- BN + ReLU + 1x1 conv (2 outputs) ----------------
__global__ void k_headout(const float* __restrict__ h0, const float* __restrict__ h1,
                          const float* __restrict__ stats,
                          const float* g0, const float* be0, const float* w20, const float* b20,
                          const float* g1, const float* be1, const float* w21, const float* b21,
                          float* outp){
  __shared__ float r0[128], r1[128];
  int t = threadIdx.x;
  int head = blockIdx.x / (BB*NPIX);
  int bn = blockIdx.x % (BB*NPIX);
  const float* h  = head ? h1  : h0;
  const float* gg = head ? g1  : g0;
  const float* bb = head ? be1 : be0;
  const float* w2 = head ? w21 : w20;
  const float* b2 = head ? b21 : b20;
  float v = h[(size_t)bn*OCH + t];
  float mu = stats[head*256 + t], va = stats[head*256 + 128 + t];
  float z = (v - mu) * rsqrtf(va + 1e-5f) * gg[t] + bb[t];
  z = fmaxf(z, 0.f);
  r0[t] = z * w2[t]; r1[t] = z * w2[128 + t];
  __syncthreads();
  for (int o = 64; o > 0; o >>= 1){ if (t < o){ r0[t] += r0[t+o]; r1[t] += r1[t+o]; } __syncthreads(); }
  if (t == 0){
    int b = bn / NPIX, n = bn % NPIX;
    size_t base = (size_t)4718592 + (size_t)head * 18432;
    outp[base + ((size_t)b*2 + 0)*NPIX + n] = r0[0] + b2[0];
    outp[base + ((size_t)b*2 + 1)*NPIX + n] = r1[0] + b2[1];
  }
}

extern "C" void kernel_launch(void* const* d_in, const int* in_sizes, int n_in,
                              void* d_out, int out_size, void* d_ws, size_t ws_size,
                              hipStream_t stream) {
  const float* img  = (const float*)d_in[0];
  const float* txt  = (const float*)d_in[1];
  const float* dens = (const float*)d_in[2];
  const float* tpw  = (const float*)d_in[3];
  const float* tpb  = (const float*)d_in[4];
  const float* wv_  = (const float*)d_in[9];
  const float* bv_  = (const float*)d_in[10];
  const float* wo_  = (const float*)d_in[11];
  const float* bo_  = (const float*)d_in[12];
  const float* lng  = (const float*)d_in[13];
  const float* lnb  = (const float*)d_in[14];
  const float* opw  = (const float*)d_in[15];
  const float* opb  = (const float*)d_in[16];
  const float* h1w  = (const float*)d_in[17];
  const float* h1b  = (const float*)d_in[18];
  const float* hbg  = (const float*)d_in[19];
  const float* hbb  = (const float*)d_in[20];
  const float* h2w  = (const float*)d_in[21];
  const float* h2b  = (const float*)d_in[22];
  const float* v1w  = (const float*)d_in[23];
  const float* v1b  = (const float*)d_in[24];
  const float* vbg  = (const float*)d_in[25];
  const float* vbb  = (const float*)d_in[26];
  const float* v2w  = (const float*)d_in[27];
  const float* v2b  = (const float*)d_in[28];
  const float* temp = (const float*)d_in[29];
  const float* sw   = (const float*)d_in[30];
  float* outp = (float*)d_out;

  // workspace layout (bytes)
  const size_t SZ_COST = (size_t)BB*NPIX*NPIX*4;   // 169,869,312
  const size_t SZ_TT   = (size_t)BB*NPIX*NPIX*2;   //  84,934,656
  const size_t SZ_BNC2 = (size_t)BB*NPIX*CC*2;     //   9,437,184
  const size_t SZ_H    = (size_t)BB*NPIX*OCH*4;    //   9,437,184
  char* w = (char*)d_ws;
  size_t off = 0;
  float*          costB = (float*)(w + off);          size_t OFF_COST = off; off += SZ_COST;
  unsigned short* TtB   = (unsigned short*)(w + off); off += SZ_TT;
  unsigned short* srchB = (unsigned short*)(w + off); off += SZ_BNC2;
  unsigned short* sftB  = (unsigned short*)(w + off); off += SZ_BNC2;
  unsigned short* fusedB= (unsigned short*)(w + off); off += SZ_BNC2;
  unsigned short* outhB = (unsigned short*)(w + off); off += SZ_BNC2;
  float*          h0B   = (float*)(w + off);          off += SZ_H;
  float*          h1B   = (float*)(w + off);          off += SZ_H;
  float*          cosB  = (float*)(w + off);          off += (size_t)BB*NPIX*4;
  float*          logpB = (float*)(w + off);          off += (size_t)BB*NPIX*4;
  float*          logqB = (float*)(w + off);          off += (size_t)BB*NPIX*4;
  float*          fB    = (float*)(w + off);          off += (size_t)BB*NPIX*4;
  float*          gB    = (float*)(w + off);          off += (size_t)BB*NPIX*4;
  unsigned short* opwhB = (unsigned short*)(w + off); off += (size_t)CC*CC*2;
  unsigned short* w1hB  = (unsigned short*)(w + off); off += (size_t)OCH*KHEAD*2;
  unsigned short* v1hB  = (unsigned short*)(w + off); off += (size_t)OCH*KHEAD*2;
  float*          avecB = (float*)(w + off);          off += (size_t)BB*CC*4;
  float*          txtnB = (float*)(w + off);          off += (size_t)BB*CC*4;
  float*          bnB   = (float*)(w + off);          off += 2048;
  if (ws_size < off) return;
  // im2col buffer aliases the cost matrix (cost is dead after k_T)
  unsigned short* A2B = (unsigned short*)(w + OFF_COST);

  k_text<<<1, 256, 0, stream>>>(txt, tpw, tpb, wv_, bv_, wo_, bo_, avecB, txtnB);
  k_convw<<<1152, 256, 0, stream>>>(opw, h1w, v1w, opwhB, w1hB, v1hB);
  k_src<<<72, 256, 0, stream>>>(img, avecB, lng, lnb, txtnB, sftB, srchB, cosB);
  k_pq<<<8, 256, 0, stream>>>(cosB, dens, temp, logpB, logqB, fB, gB);
  k_cost<<<20736, 256, 0, stream>>>(srchB, sw, costB);
  for (int it = 0; it < 3; ++it){
    k_row<<<BB*NPIX, 256, 0, stream>>>(costB, gB, logpB, fB);
    k_col<<<288, 256, 0, stream>>>(costB, fB, logqB, gB);
  }
  k_T<<<41472, 256, 0, stream>>>(costB, fB, gB, TtB);
  k_fused<<<2304, 256, 0, stream>>>(TtB, sftB, fusedB);
  k_out<<<2304, 256, 0, stream>>>(fusedB, opwhB, img, opb, outp, outhB);
  k_im2col<<<20736, 256, 0, stream>>>(outhB, A2B);
  k_headgemm<<<1152, 256, 0, stream>>>(A2B, w1hB, h1b, h0B);
  k_headgemm<<<1152, 256, 0, stream>>>(A2B, v1hB, v1b, h1B);
  k_bn<<<256, 256, 0, stream>>>(h0B, h1B, bnB);
  k_headout<<<2*BB*NPIX, 128, 0, stream>>>(h0B, h1B, bnB,
                                           hbg, hbb, h2w, h2b,
                                           vbg, vbb, v2w, v2b, outp);
}
